// VectorizedLinearAttention_60971355734466
// MI455X (gfx1250) — compile-verified, matrix operand order checked by reading, not run on hardware
//
#include <hip/hip_runtime.h>

typedef __attribute__((ext_vector_type(16))) _Float16 v16h;
typedef __attribute__((ext_vector_type(8)))  _Float16 v8h;
typedef __attribute__((ext_vector_type(4)))  _Float16 v4h;
typedef __attribute__((ext_vector_type(8)))  float    v8f;
typedef __attribute__((ext_vector_type(4)))  float    v4f;

#define N_ 2
#define T_ 8
#define L_ 2048
#define H_ 8
#define D_ 64
#define NTH (N_*T_*H_)     /* 128 slices */
#define SPLIT 8
#define ROWS1 (L_/SPLIT)   /* 256 rows per phase-1 block */
#define HD (H_*D_)         /* 512: row stride (floats) inside [N,T,L,H,D] */
#define EPS_ 1e-6f

// elu(x)+1 == x+1 (x>0) else exp(x)
__device__ __forceinline__ float fmap(float x) { return x > 0.0f ? x + 1.0f : __expf(x); }

// Assemble a 16x16x32 WMMA fragment from two contiguous 8-half (16B) LDS runs.
// Per ISA 7.12.2 (16-bit A/B): element e<8 -> K = half*8+e ; e>=8 -> K = 16+half*8+(e-8)
__device__ __forceinline__ v16h frag16(const _Float16* p0, const _Float16* p1) {
  v8h lo = *(const v8h*)p0;
  v8h hi = *(const v8h*)p1;
  v16h r;
#pragma unroll
  for (int i = 0; i < 8; ++i) { r[i] = lo[i]; r[8 + i] = hi[i]; }
  return r;
}

__global__ void zero_ws_kernel(float* __restrict__ p, int n) {
  int i = blockIdx.x * blockDim.x + threadIdx.x;
  if (i < n) p[i] = 0.0f;
}

// Phase 1: KV[slice][m][d] += K'^T V over a 256-row L-chunk; Ksum[slice][d] += col-sums of K'.
__global__ __launch_bounds__(256)
void la_kv_ksum(const float* __restrict__ Kg, const float* __restrict__ Vg,
                const float* __restrict__ klen,
                float* __restrict__ gKV, float* __restrict__ gKsum)
{
  __shared__ __align__(16) _Float16 sKt[64][40]; // [d][s], 80B row stride (16B aligned)
  __shared__ __align__(16) _Float16 sVt[64][40]; // [m][s]
  __shared__ float sKsum[64];

  const int blk   = blockIdx.x;
  const int slice = blk / SPLIT;        // (n*T+t)*H + h
  const int part  = blk % SPLIT;
  const int nt    = slice >> 3;
  const int h     = slice & 7;
  const int n     = nt / T_;
  const long base   = (long)nt * L_ * HD + (long)h * D_;
  const long klbase = (long)n * L_;

  const int tid  = threadIdx.x;
  const int lane = tid & 31;
  const int w    = tid >> 5;            // wave 0..7
  const int half = lane >> 4;
  const int l15  = lane & 15;
  const int di   = w >> 1;              // d-tile of KV this wave owns
  const int mj0  = (w & 1) * 2;         // first of two m-tiles

  // loader mapping: each thread owns 4 contiguous d columns at one s row
  const int sL = tid >> 4;              // 0..15
  const int d0 = (tid & 15) * 4;

  if (tid < 64) sKsum[tid] = 0.0f;

  v8f acc0 = {}; v8f acc1 = {};
  float kacc4[4] = {0.0f, 0.0f, 0.0f, 0.0f};

  const float* Kb = Kg + base;          // per-block base; offsets stay 32-bit
  const float* Vb = Vg + base;
  const int s_base0 = part * ROWS1;

  for (int sub = 0; sub < ROWS1 / 32; ++sub) {
    const int sbase = s_base0 + sub * 32;
    const unsigned off0 = (unsigned)(sbase + sL) * HD + d0;
#pragma unroll
    for (int p = 0; p < 2; ++p) {       // two 16-row halves of the 32-row panel
      const int s = sL + p * 16;
      const unsigned off = off0 + (unsigned)(p * 16 * HD); // +32KB imm for p=1
      const v4f k4 = *(const v4f*)(Kb + off);              // global_load_b128
      const v4f v4 = *(const v4f*)(Vb + off);              // global_load_b128
      const float m = klen[klbase + sbase + s];
#pragma unroll
      for (int j = 0; j < 4; ++j) {
        const float kv = fmap(k4[j]) * m;
        kacc4[j] += kv;
        sKt[d0 + j][s] = (_Float16)kv;   // transposed scatter (b16 stores)
        sVt[d0 + j][s] = (_Float16)v4[j];
      }
    }
    if (sub + 1 < ROWS1 / 32) {          // pull next panel toward L0/L2
      __builtin_prefetch(Kb + off0 + 32u * HD, 0, 3);
      __builtin_prefetch(Vb + off0 + 32u * HD, 0, 3);
    }
    __syncthreads();

    const int dA = di * 16 + l15;
    v16h a  = frag16(&sKt[dA][half * 8], &sKt[dA][16 + half * 8]);
    const int m0 = mj0 * 16 + l15;
    v16h b0 = frag16(&sVt[m0][half * 8], &sVt[m0][16 + half * 8]);
    v16h b1 = frag16(&sVt[m0 + 16][half * 8], &sVt[m0 + 16][16 + half * 8]);
    acc0 = __builtin_amdgcn_wmma_f32_16x16x32_f16(false, a, false, b0, (short)0, acc0, false, false);
    acc1 = __builtin_amdgcn_wmma_f32_16x16x32_f16(false, a, false, b1, (short)0, acc1, false, false);
    __syncthreads();
  }

  // Ksum: block-local LDS reduce, then one global atomic per d
#pragma unroll
  for (int j = 0; j < 4; ++j) atomicAdd(&sKsum[d0 + j], kacc4[j]);
  __syncthreads();
  if (tid < 64) atomicAdd(&gKsum[slice * 64 + tid], sKsum[tid]);

  // KV partials: D-frag row = d_local (A's M), col = m (B's N)
  const long kvbase = (long)slice * 4096;
#pragma unroll
  for (int r = 0; r < 8; ++r) {
    const int drow = di * 16 + r + half * 8;
    atomicAdd(&gKV[kvbase + (long)(mj0 * 16 + l15) * 64 + drow], acc0[r]);
    atomicAdd(&gKV[kvbase + (long)((mj0 + 1) * 16 + l15) * 64 + drow], acc1[r]);
  }
}

// Phase 2: out[l,m] = (Q'[l,:] . KV[m,:]) * Z[l]
__global__ __launch_bounds__(256)
void la_out(const float* __restrict__ Qg, const float* __restrict__ gKV,
            const float* __restrict__ gKsum, float* __restrict__ Og)
{
  __shared__ __align__(16) _Float16 sQ[64][72];   // [l][d], 144B row stride
  __shared__ __align__(16) _Float16 sKV[64][72];  // [m][d]
  __shared__ float zbuf[64];
  __shared__ float sKs[64];

  const int blk    = blockIdx.x;
  const int slice  = blk >> 5;
  const int rowblk = blk & 31;
  const int nt = slice >> 3;
  const int h  = slice & 7;
  const long base = (long)nt * L_ * HD + (long)h * D_;
  const int l0 = rowblk * 64;

  const int tid  = threadIdx.x;
  const int lane = tid & 31;
  const int w    = tid >> 5;
  const int half = lane >> 4;
  const int l15  = lane & 15;
  const int ri   = w >> 1;              // row tile
  const int mj0  = (w & 1) * 2;

  const int lL = tid >> 4;              // 0..15
  const int d0 = (tid & 15) * 4;

  if (tid < 64) sKs[tid] = gKsum[slice * 64 + tid];

  const float* Qb  = Qg + base + (long)l0 * HD;
  const float* KVb = gKV + (long)slice * 4096;
#pragma unroll
  for (int p = 0; p < 4; ++p) {
    const int l = lL + p * 16;
    const v4f q4  = *(const v4f*)(Qb + (unsigned)l * HD + d0);  // global_load_b128
    const v4f kv4 = *(const v4f*)(KVb + l * 64 + d0);           // global_load_b128
    v4h qh, kh;
#pragma unroll
    for (int j = 0; j < 4; ++j) {
      qh[j] = (_Float16)fmap(q4[j]);
      kh[j] = (_Float16)kv4[j];
    }
    *(v4h*)&sQ[l][d0]  = qh;            // one ds_store_b64 each
    *(v4h*)&sKV[l][d0] = kh;
  }
  __syncthreads();

  if (tid < 64) {                        // per-row normalizer in f32
    float zt = 0.0f;
#pragma unroll
    for (int d = 0; d < 64; ++d) zt += (float)sQ[tid][d] * sKs[d];
    zbuf[tid] = 1.0f / (zt + EPS_);
  }
  __syncthreads();

  v8f acc0 = {}; v8f acc1 = {};
  const int lrow = ri * 16 + l15;
  const int m0   = mj0 * 16 + l15;
#pragma unroll
  for (int kc = 0; kc < 2; ++kc) {       // D=64 -> two K=32 chunks
    const int o = kc * 32;
    v16h a  = frag16(&sQ[lrow][o + half * 8],      &sQ[lrow][o + 16 + half * 8]);
    v16h b0 = frag16(&sKV[m0][o + half * 8],       &sKV[m0][o + 16 + half * 8]);
    v16h b1 = frag16(&sKV[m0 + 16][o + half * 8],  &sKV[m0 + 16][o + 16 + half * 8]);
    acc0 = __builtin_amdgcn_wmma_f32_16x16x32_f16(false, a, false, b0, (short)0, acc0, false, false);
    acc1 = __builtin_amdgcn_wmma_f32_16x16x32_f16(false, a, false, b1, (short)0, acc1, false, false);
  }

#pragma unroll
  for (int r = 0; r < 8; ++r) {
    const int row = ri * 16 + r + half * 8;
    const float z = zbuf[row];
    const long go = base + (long)(l0 + row) * HD;
    Og[go + mj0 * 16 + l15]       = acc0[r] * z;
    Og[go + (mj0 + 1) * 16 + l15] = acc1[r] * z;
  }
}

extern "C" void kernel_launch(void* const* d_in, const int* in_sizes, int n_in,
                              void* d_out, int out_size, void* d_ws, size_t ws_size,
                              hipStream_t stream) {
  const float* Qg   = (const float*)d_in[0];
  const float* Kg   = (const float*)d_in[1];
  const float* Vg   = (const float*)d_in[2];
  const float* klen = (const float*)d_in[3];
  float* out   = (float*)d_out;
  float* gKV   = (float*)d_ws;                      // 128*64*64 f32
  float* gKsum = gKV + (long)NTH * 64 * 64;         // 128*64 f32

  const int zn = NTH * 64 * 64 + NTH * 64;
  zero_ws_kernel<<<(zn + 255) / 256, 256, 0, stream>>>(gKV, zn);
  la_kv_ksum<<<NTH * SPLIT, 256, 0, stream>>>(Kg, Vg, klen, gKV, gKsum);
  la_out<<<NTH * 32, 256, 0, stream>>>(Qg, gKV, gKsum, out);
}
